// SynchronizationLoss_82660940579472
// MI455X (gfx1250) — matrix operand order repaired and checked by reading, compiled, not run on hardware
//
#include <hip/hip_runtime.h>
#include <hip/hip_bf16.h>

typedef __attribute__((ext_vector_type(16))) __bf16         v16bf;
typedef __attribute__((ext_vector_type(16))) unsigned short v16u;
typedef __attribute__((ext_vector_type(8)))  float          v8f;

#define N_TRIALS   8
#define T_MS       2500
#define N_NEURONS  16000
#define N_SAMPLES  50
#define K_MAX      160
#define N_BINSZ    20
#define N_TILES    157          // ceil(2500/16)
#define T_PAD      (N_TILES*16) // 2512
#define SEL_OFF    64           // float offset of sel[] in workspace (fanos live at 0)
#define EPSF       1e-7f
#define ONE_BF16   ((unsigned short)0x3F80)  // bf16 bit pattern of 1.0f

// -------------------------------------------------------------------------
// Kernel 1: sel[s, t] = sum_k spikes[trial(s), t, idx(s,k)] * (k < count(s))
// One wave32 per (sample, 16-row t tile), mapped onto v_wmma_f32_16x16x32_bf16.
//
// Trick: spike values are exactly 0.0f / 1.0f, so the bf16 representation is
// the upper half-word of the f32 bit pattern. We gather only those 16-bit
// half-words (u16 loads of word 2*idx+1) and bit-cast straight into the bf16
// A fragment -- no float->16-bit conversion ALU ops at all. The count mask
// goes into B as the bf16 literal 0x3F80, broadcast over all N columns.
// C (f32) accumulates over the 5 k-chunks; sums <= 160 are exact.
// -------------------------------------------------------------------------
__global__ __launch_bounds__(32)
void sel_wmma_kernel(const float* __restrict__ spikes,
                     const int*   __restrict__ sample_trials,
                     const int*   __restrict__ sample_idx,
                     const int*   __restrict__ sample_counts,
                     float*       __restrict__ sel)
{
    __shared__ int sidx[K_MAX];

    const int s    = blockIdx.y;
    const int tile = blockIdx.x;
    const int lane = threadIdx.x;          // 0..31, one full wave, EXEC all-1s

    // Stage this sample's neuron indices in LDS (160/32 = 5 each, no divergence)
    const int* idxrow = sample_idx + s * K_MAX;
    for (int i = lane; i < K_MAX; i += 32) sidx[i] = idxrow[i];
    const int trial = sample_trials[s];
    const int count = sample_counts[s];
    __syncthreads();

    const int m  = lane & 15;              // row M within tile (A layout)
    const int hi = lane >> 4;              // lane group 0/1
    int t  = tile * 16 + m;
    int tc = t < T_MS ? t : (T_MS - 1);    // clamp padded rows (result ignored)
    // View this (trial, t) row as u16 half-words; element i's bf16 bits live
    // at half-word 2*i + 1.
    const unsigned short* __restrict__ sp16 =
        (const unsigned short*)(spikes + ((size_t)trial * T_MS + (size_t)tc) * N_NEURONS);

    v8f c = {};
    #pragma unroll
    for (int kc = 0; kc < K_MAX; kc += 32) {
        // A fragment: ISA 16-bit A 16x32 layout.
        // lanes 0-15 hold K = {0..7, 16..23}; lanes 16-31 hold K = {8..15, 24..31}
        v16u au;
        #pragma unroll
        for (int e = 0; e < 16; ++e) {
            int kl = (e < 8 ? e : e + 8) + hi * 8;          // K within chunk
            au[e] = sp16[2 * sidx[kc + kl] + 1];            // gather bf16 bits
        }
        // B fragment: mask[k] (bf16 1.0 / 0.0) broadcast across all N columns.
        // lanes 0-15 hold K = {0..7, 16..23}; lanes 16-31 hold K = {8..15, 24..31}
        v16u bu;
        #pragma unroll
        for (int e = 0; e < 16; ++e) {
            int kb = kc + (e < 8 ? e : e + 8) + hi * 8;     // global k for this slot
            bu[e] = (kb < count) ? ONE_BF16 : (unsigned short)0;
        }
        v16bf a = __builtin_bit_cast(v16bf, au);
        v16bf b = __builtin_bit_cast(v16bf, bu);
        c = __builtin_amdgcn_wmma_f32_16x16x32_bf16(
                /*neg_a=*/false, a, /*neg_b=*/false, b,
                /*c_mod=*/(short)0, c, /*reuse_a=*/false, /*reuse_b=*/false);
    }

    // D layout: lanes 0-15 VGPR r -> M=r, lanes 16-31 -> M=8+r, N = lane&15.
    // Column N==0 (lanes 0 and 16) carries sel for the tile's 16 rows.
    if (m == 0) {
        float* out = sel + (size_t)s * T_PAD + tile * 16 + hi * 8;
        #pragma unroll
        for (int r = 0; r < 8; ++r) out[r] = c[r];
    }
}

// -------------------------------------------------------------------------
// Kernel 2: per bin size, Fano factor averaged over samples.
// One block per bin size; two-pass mean/variance to match jnp.var numerics.
// -------------------------------------------------------------------------
__global__ __launch_bounds__(256)
void fano_kernel(const float* __restrict__ sel, float* __restrict__ fanos)
{
    static const int BS[N_BINSZ] = {1, 1, 2, 3, 4, 6, 9, 13, 18, 26,
                                    38, 55, 78, 113, 162, 234, 336, 483, 695, 1000};
    __shared__ float red[256];
    __shared__ float mean_sh;
    __shared__ float facc;

    const int b   = blockIdx.x;
    const int tid = threadIdx.x;
    const int bs  = BS[b];
    const int nb  = T_MS / bs;

    if (tid == 0) facc = 0.0f;
    __syncthreads();

    for (int s = 0; s < N_SAMPLES; ++s) {
        const float* row = sel + (size_t)s * T_PAD;

        // pass 1: mean of binned counts
        float psum = 0.0f;
        for (int j = tid; j < nb; j += 256) {
            float cb = 0.0f;
            const float* p = row + j * bs;
            for (int u = 0; u < bs; ++u) cb += p[u];
            psum += cb;
        }
        red[tid] = psum;
        __syncthreads();
        for (int off = 128; off > 0; off >>= 1) {
            if (tid < off) red[tid] += red[tid + off];
            __syncthreads();
        }
        if (tid == 0) mean_sh = red[0] / (float)nb;
        __syncthreads();
        const float mean = mean_sh;

        // pass 2: variance (population, ddof=0)
        float pvar = 0.0f;
        for (int j = tid; j < nb; j += 256) {
            float cb = 0.0f;
            const float* p = row + j * bs;
            for (int u = 0; u < bs; ++u) cb += p[u];
            float d = cb - mean;
            pvar += d * d;
        }
        red[tid] = pvar;
        __syncthreads();
        for (int off = 128; off > 0; off >>= 1) {
            if (tid < off) red[tid] += red[tid + off];
            __syncthreads();
        }
        if (tid == 0) {
            float var = red[0] / (float)nb;
            facc += var / fmaxf(mean, EPSF);
        }
        __syncthreads();
    }
    if (tid == 0) fanos[b] = facc / (float)N_SAMPLES;
}

// -------------------------------------------------------------------------
// Kernel 3: loss = 10 * mean((exp_fanos - fanos)^2)
// -------------------------------------------------------------------------
__global__ __launch_bounds__(32)
void loss_kernel(const float* __restrict__ fanos,
                 const float* __restrict__ exp_fanos,
                 float*       __restrict__ out)
{
    if (threadIdx.x == 0 && blockIdx.x == 0) {
        float acc = 0.0f;
        for (int i = 0; i < N_BINSZ; ++i) {
            float d = exp_fanos[i] - fanos[i];
            acc += d * d;
        }
        out[0] = 10.0f * (acc / (float)N_BINSZ);
    }
}

extern "C" void kernel_launch(void* const* d_in, const int* in_sizes, int n_in,
                              void* d_out, int out_size, void* d_ws, size_t ws_size,
                              hipStream_t stream)
{
    (void)in_sizes; (void)n_in; (void)out_size; (void)ws_size;

    const float* spikes        = (const float*)d_in[0]; // (8, 2500, 16000)
    const float* exp_fanos     = (const float*)d_in[1]; // (20,)
    const int*   sample_trials = (const int*)  d_in[2]; // (50,)
    const int*   sample_idx    = (const int*)  d_in[3]; // (50, 160)
    const int*   sample_counts = (const int*)  d_in[4]; // (50,)
    float*       out           = (float*)d_out;         // scalar

    float* wsf   = (float*)d_ws;
    float* fanos = wsf;             // [0, 20)
    float* sel   = wsf + SEL_OFF;   // 50 * 2512 floats (~0.5 MB)

    dim3 grid1(N_TILES, N_SAMPLES);
    sel_wmma_kernel<<<grid1, 32, 0, stream>>>(spikes, sample_trials, sample_idx,
                                              sample_counts, sel);
    fano_kernel<<<N_BINSZ, 256, 0, stream>>>(sel, fanos);
    loss_kernel<<<1, 32, 0, stream>>>(fanos, exp_fanos, out);
}
